// InfiniAttention_27230092657393
// MI455X (gfx1250) — compile-verified, matrix operand order checked by reading, not run on hardware
//
#include <hip/hip_runtime.h>
#include <hip/hip_bf16.h>

typedef __bf16 bf16_t;
typedef __attribute__((ext_vector_type(16))) __bf16 v16bf;
typedef __attribute__((ext_vector_type(8)))  float  v8f;
typedef __attribute__((ext_vector_type(4)))  unsigned int v4u;
typedef __attribute__((ext_vector_type(8)))  int  v8i;
typedef __attribute__((ext_vector_type(4)))  int  v4i;

#define WMMA_BF16(a, b, c) \
  __builtin_amdgcn_wmma_f32_16x16x32_bf16(false, (a), false, (b), (short)0, (c), false, false)

#if defined(__has_builtin)
#  if __has_builtin(__builtin_amdgcn_tensor_load_to_lds)
#    define HAS_TDM 1
#  endif
#endif
#ifndef HAS_TDM
#  define HAS_TDM 0
#endif

// ---------------------------------------------------------------------------
// Packed-tile convention: a 16x32 (A) or 32x16 (B) bf16 tile is stored as 512
// contiguous bf16: element (lane, e) at [lane*16 + e], matching the CDNA5 WMMA
// VGPR layout (ISA §7.12.2): m|n = lane&15, k = kpair(e>>1, lane>>4) + (e&1).
// Fragment load = 32 contiguous bytes per lane -> global/ds _load_b128 pairs.
// ---------------------------------------------------------------------------

static __device__ __forceinline__ void fwd_map(int w, int& mn, int& k) {
  const int lane = w >> 4, e = w & 15;
  const int half = lane >> 4, v = e >> 1, lo = e & 1;
  mn = lane & 15;
  k = ((v < 4) ? 2 * v : 2 * v + 8) + 8 * half + lo;
}

static __device__ __forceinline__ int inv_addr(int k) {
  const int lo = k & 1, q = k >> 1;
  const int hr = (q >> 2) & 1;
  const int vr = (q & 3) + 4 * (q >> 3);
  return hr * 256 + 2 * vr + lo;
}

#if HAS_TDM
// TDM 2D tile load: rows x cols (bf16 elements), row stride in elements.
// Descriptor per CDNA5 ISA §8: group0 {count, lds_addr, global_addr, type=2},
// group1 {data_size=2B, tensor dims (no-OOB), tile dims, dim0 stride}.
static __device__ __forceinline__ void tdm_load_2d(unsigned lds_addr,
                                                   const void* gptr,
                                                   unsigned tile_d0,
                                                   unsigned tile_d1,
                                                   unsigned long long stride0) {
  const unsigned long long ga = (unsigned long long)gptr;
  v4u g0;
  g0[0] = 1u;                                            // count=1, user mode
  g0[1] = lds_addr;                                      // LDS byte address
  g0[2] = (unsigned)(ga & 0xffffffffu);                  // global_addr[31:0]
  g0[3] = (unsigned)((ga >> 32) & 0x1ffffffu) | (2u << 30);  // [56:32] | type=2
  const unsigned td0 = 0x7fffffffu, td1 = 0x7fffffffu;   // tensor dims: no OOB
  v8i g1;
  g1[0] = (int)(1u << 16);                               // data_size=1 (2 bytes)
  g1[1] = (int)((td0 & 0xffffu) << 16);                  // tensor_dim0[15:0]
  g1[2] = (int)((td0 >> 16) | ((td1 & 0xffffu) << 16));  // dim0 hi | dim1 lo
  g1[3] = (int)((td1 >> 16) | (tile_d0 << 16));          // dim1 hi | tile_dim0
  g1[4] = (int)(tile_d1 & 0xffffu);                      // tile_dim1 (dim2=0)
  g1[5] = (int)(stride0 & 0xffffffffu);                  // dim0_stride[31:0]
  g1[6] = (int)((stride0 >> 32) & 0xffffu);              // dim0_stride[47:32]
  g1[7] = 0;
  v4i z4 = {0, 0, 0, 0};
#if defined(__clang_major__) && (__clang_major__ >= 23)
  v8i z8 = {0, 0, 0, 0, 0, 0, 0, 0};
  __builtin_amdgcn_tensor_load_to_lds(g0, g1, z4, z4, z8, 0);
#else
  __builtin_amdgcn_tensor_load_to_lds(g0, g1, z4, z4, 0);
#endif
}
#endif

// ---- pack kernels: f32 row-major -> packed bf16 fragment tiles ----

__global__ void pack_x(const float* __restrict__ X, bf16_t* __restrict__ P) {
  for (int p = blockIdx.x * blockDim.x + threadIdx.x; p < 2097152;
       p += gridDim.x * blockDim.x) {
    const int w = p & 511, tile = p >> 9;
    const int kt = tile & 31, mt = tile >> 5;
    int m, k; fwd_map(w, m, k);
    P[p] = (bf16_t)X[(mt * 16 + m) * 1024 + kt * 32 + k];
  }
}

__global__ void pack_w(const float* __restrict__ W, bf16_t* __restrict__ P) {
  for (int p = blockIdx.x * blockDim.x + threadIdx.x; p < 1048576;
       p += gridDim.x * blockDim.x) {
    const int w = p & 511, tile = p >> 9;
    const int kt = tile & 31, nt = tile >> 5;
    int n, k; fwd_map(w, n, k);
    P[p] = (bf16_t)W[(kt * 32 + k) * 1024 + nt * 16 + n];
  }
}

__global__ void pack_q(const float* __restrict__ Qf, bf16_t* __restrict__ P) {
  for (int p = blockIdx.x * blockDim.x + threadIdx.x; p < 2097152;
       p += gridDim.x * blockDim.x) {
    const int w = p & 511, tile = p >> 9;
    const int kft = tile & 1, t2 = tile >> 1;
    const int h = t2 & 15, mt = t2 >> 4;
    int m, k; fwd_map(w, m, k);
    P[p] = (bf16_t)(Qf[(mt * 16 + m) * 1024 + h * 64 + kft * 32 + k] * 0.125f);
  }
}

__global__ void pack_kscore(const float* __restrict__ Kf, bf16_t* __restrict__ P) {
  for (int p = blockIdx.x * blockDim.x + threadIdx.x; p < 2097152;
       p += gridDim.x * blockDim.x) {
    const int w = p & 511, tile = p >> 9;
    const int kft = tile & 1, t2 = tile >> 1;
    const int ktile = t2 & 127, h = t2 >> 7;
    int n, k; fwd_map(w, n, k);
    P[p] = (bf16_t)Kf[(ktile * 16 + n) * 1024 + h * 64 + kft * 32 + k];
  }
}

__global__ void pack_v(const float* __restrict__ Vf, bf16_t* __restrict__ P) {
  for (int p = blockIdx.x * blockDim.x + threadIdx.x; p < 2097152;
       p += gridDim.x * blockDim.x) {
    const int w = p & 511, tile = p >> 9;
    const int nt = tile & 3, t2 = tile >> 2;
    const int kc = t2 & 63, h = t2 >> 6;
    int n, k; fwd_map(w, n, k);
    P[p] = (bf16_t)Vf[(kc * 32 + k) * 1024 + h * 64 + nt * 16 + n];
  }
}

// ---- bf16 WMMA GEMM on packed tiles: C[2048x1024] = A * W ----
// 8 waves/block share the B panel: one wave streams it into a double-buffered
// LDS tile with the Tensor Data Mover (TENSORcnt + barrier sync); each wave
// computes a 16x64 slab reading B fragments as contiguous ds_load_b128.
__global__ void __launch_bounds__(256)
wmma_gemm(const bf16_t* __restrict__ Ap, const bf16_t* __restrict__ Wp,
          float* __restrict__ Of) {
  __shared__ bf16_t Bst[2][2048];            // 2 x (4 tiles x 512 bf16) = 8 KiB
  const int lane = threadIdx.x & 31, wave = threadIdx.x >> 5;
  const int half = lane >> 4, nn = lane & 15;
  const int mt = blockIdx.x * 8 + wave;
  const int nt0 = blockIdx.y * 4;
  const bf16_t* ap   = Ap + (size_t)(mt * 32) * 512 + lane * 16;
  const bf16_t* wsrc = Wp + (size_t)(nt0 * 32) * 512;   // + kt*512; row stride 16384
  v8f o0 = {}, o1 = {}, o2 = {}, o3 = {};

#if HAS_TDM
  if (wave == 0) {
    tdm_load_2d((unsigned)(unsigned long long)(void*)&Bst[0][0], wsrc, 512u, 4u,
                16384ull);
    __builtin_amdgcn_s_wait_tensorcnt(0);
  }
  __syncthreads();
  for (int kt = 0; kt < 32; ++kt) {
    if (wave == 0 && kt + 1 < 32)            // async: stream next B chunk
      tdm_load_2d((unsigned)(unsigned long long)(void*)&Bst[(kt + 1) & 1][0],
                  wsrc + (kt + 1) * 512, 512u, 4u, 16384ull);
    __builtin_prefetch(ap + kt * 512 + 512, 0, 1);
    const bf16_t* bs = &Bst[kt & 1][lane * 16];
    v16bf a  = *(const v16bf*)(ap + kt * 512);
    v16bf b0 = *(const v16bf*)(bs);
    v16bf b1 = *(const v16bf*)(bs + 512);
    v16bf b2 = *(const v16bf*)(bs + 1024);
    v16bf b3 = *(const v16bf*)(bs + 1536);
    o0 = WMMA_BF16(a, b0, o0);
    o1 = WMMA_BF16(a, b1, o1);
    o2 = WMMA_BF16(a, b2, o2);
    o3 = WMMA_BF16(a, b3, o3);
    if (kt + 1 < 32) {
      if (wave == 0) __builtin_amdgcn_s_wait_tensorcnt(0);
      __syncthreads();
    }
  }
#else
  for (int kt = 0; kt < 32; ++kt) {
    __syncthreads();
    {   // cooperative copy of the 4-tile B chunk (2048 bf16), 16B per thread
      const int idx = threadIdx.x * 8;
      const int row = idx >> 9, col = idx & 511;
      *(uint4*)(&Bst[kt & 1][idx]) =
          *(const uint4*)(wsrc + kt * 512 + row * 16384 + col);
    }
    __syncthreads();
    const bf16_t* bs = &Bst[kt & 1][lane * 16];
    v16bf a  = *(const v16bf*)(ap + kt * 512);
    v16bf b0 = *(const v16bf*)(bs);
    v16bf b1 = *(const v16bf*)(bs + 512);
    v16bf b2 = *(const v16bf*)(bs + 1024);
    v16bf b3 = *(const v16bf*)(bs + 1536);
    o0 = WMMA_BF16(a, b0, o0);
    o1 = WMMA_BF16(a, b1, o1);
    o2 = WMMA_BF16(a, b2, o2);
    o3 = WMMA_BF16(a, b3, o3);
  }
#endif

#pragma unroll
  for (int v = 0; v < 8; ++v) {
    const int row = mt * 16 + v + 8 * half;
    float vals[4] = {o0[v], o1[v], o2[v], o3[v]};
#pragma unroll
    for (int t = 0; t < 4; ++t)
      Of[row * 1024 + (nt0 + t) * 16 + nn] = vals[t];
  }
}

// ---- fused causal flash attention + compressive-memory retrieval + gate ----
__global__ void __launch_bounds__(256)
attn_kernel(const bf16_t* __restrict__ Qp, const bf16_t* __restrict__ Kp,
            const bf16_t* __restrict__ Vp, const float* __restrict__ Qf,
            const float* __restrict__ Min, const float* __restrict__ zin,
            const float* __restrict__ beta, bf16_t* __restrict__ Op) {
  __shared__ float  Mlds[4096];
  __shared__ float  zlds[64];
  __shared__ bf16_t Pl[8 * 512];   // per-wave P tile, stored in packed-A order
  const int h = blockIdx.y;
  const int tid = threadIdx.x;
  for (int i = tid; i < 4096; i += 256) Mlds[i] = Min[h * 4096 + i];
  if (tid < 64) zlds[tid] = zin[h * 64 + tid];
  __syncthreads();

  const int lane = tid & 31, wave = tid >> 5;
  const int half = lane >> 4, nn = lane & 15;
  const int q0 = blockIdx.x * 128 + wave * 16;
  const int ce = inv_addr(nn);
  const int co = inv_addr(16 + nn);

  const bf16_t* qp = Qp + ((size_t)(((q0 >> 4) * 16 + h) * 2) << 9) + lane * 16;
  const v16bf qA0 = *(const v16bf*)(qp);
  const v16bf qA1 = *(const v16bf*)(qp + 512);
  bf16_t* Pw = Pl + wave * 512;

  v8f O0 = {}, O1 = {}, O2 = {}, O3 = {};
  float rm[8], rs[8];
#pragma unroll
  for (int v = 0; v < 8; ++v) { rm[v] = -1e30f; rs[v] = 0.f; }

  const int nkb = (q0 + 16 + 31) >> 5;   // causal: keys <= q0+15
  for (int kb = 0; kb < nkb; ++kb) {
    const int kbase = kb * 32;
    const bf16_t* kp = Kp + ((size_t)((h * 128 + kb * 2) * 2) << 9) + lane * 16;
    v8f c0 = {}, c1 = {};
    c0 = WMMA_BF16(qA0, *(const v16bf*)(kp),        c0);
    c0 = WMMA_BF16(qA1, *(const v16bf*)(kp + 512),  c0);
    c1 = WMMA_BF16(qA0, *(const v16bf*)(kp + 1024), c1);
    c1 = WMMA_BF16(qA1, *(const v16bf*)(kp + 1536), c1);

    float bm[8];
#pragma unroll
    for (int v = 0; v < 8; ++v) {
      const int row = q0 + v + 8 * half;
      float s0 = (kbase + nn > row)      ? -1e30f : c0[v];
      float s1 = (kbase + 16 + nn > row) ? -1e30f : c1[v];
      c0[v] = s0; c1[v] = s1;
      bm[v] = fmaxf(s0, s1);
    }
#pragma unroll
    for (int m = 1; m < 16; m <<= 1) {
#pragma unroll
      for (int v = 0; v < 8; ++v) bm[v] = fmaxf(bm[v], __shfl_xor(bm[v], m, 32));
    }
    float bs[8];
#pragma unroll
    for (int v = 0; v < 8; ++v) {
      const float mnew  = fmaxf(rm[v], bm[v]);
      const float alpha = __expf(rm[v] - mnew);
      const float p0 = __expf(c0[v] - mnew);
      const float p1 = __expf(c1[v] - mnew);
      rm[v] = mnew;
      bs[v] = p0 + p1;
      O0[v] *= alpha; O1[v] *= alpha; O2[v] *= alpha; O3[v] *= alpha;
      rs[v] *= alpha;
      const int rr16 = (v + 8 * half) * 16;
      Pw[ce + rr16] = (bf16_t)p0;
      Pw[co + rr16] = (bf16_t)p1;
    }
#pragma unroll
    for (int m = 1; m < 16; m <<= 1) {
#pragma unroll
      for (int v = 0; v < 8; ++v) bs[v] += __shfl_xor(bs[v], m, 32);
    }
#pragma unroll
    for (int v = 0; v < 8; ++v) rs[v] += bs[v];

    const v16bf pA = *(const v16bf*)(Pw + lane * 16);
    const bf16_t* vp = Vp + ((size_t)((h * 64 + kb) * 4) << 9) + lane * 16;
    O0 = WMMA_BF16(pA, *(const v16bf*)(vp),        O0);
    O1 = WMMA_BF16(pA, *(const v16bf*)(vp + 512),  O1);
    O2 = WMMA_BF16(pA, *(const v16bf*)(vp + 1024), O2);
    O3 = WMMA_BF16(pA, *(const v16bf*)(vp + 1536), O3);
  }

  const float g = 1.f / (1.f + __expf(-beta[h]));
  bf16_t* obase = Op + ((size_t)((q0 >> 4) * 32 + h * 2) << 9);
#pragma unroll
  for (int v = 0; v < 8; ++v) {
    const int rr = v + 8 * half;
    const int row = q0 + rr;
    const float inv = 1.f / rs[v];
    const float* qr = Qf + row * 1024 + h * 64;
    float den = 0.f, num0 = 0.f, num1 = 0.f, num2 = 0.f, num3 = 0.f;
    for (int j = 0; j < 64; ++j) {
      const float qv = qr[j];
      const float s = qv > 0.f ? qv + 1.f : __expf(qv);  // elu(x)+1
      den += s * zlds[j];
      const float* Mr = Mlds + j * 64;
      num0 += s * Mr[nn];      num1 += s * Mr[16 + nn];
      num2 += s * Mr[32 + nn]; num3 += s * Mr[48 + nn];
    }
    const float dinv = 1.f / (den + 1e-6f);
    const float og = 1.f - g;
    obase[ce + rr * 16]       = (bf16_t)(g * num0 * dinv + og * O0[v] * inv);
    obase[co + rr * 16]       = (bf16_t)(g * num1 * dinv + og * O1[v] * inv);
    obase[512 + ce + rr * 16] = (bf16_t)(g * num2 * dinv + og * O2[v] * inv);
    obase[512 + co + rr * 16] = (bf16_t)(g * num3 * dinv + og * O3[v] * inv);
  }
}

// ---- M_new = M + sum_s sigma(k)^T v, full f32 (state outputs) ----
__global__ void __launch_bounds__(256)
mem_update(const float* __restrict__ Kf, const float* __restrict__ Vf,
           const float* __restrict__ Min, float* __restrict__ Mout) {
  __shared__ float skc[1024];
  __shared__ float vc[1024];
  const int h = blockIdx.y;
  const int ti = blockIdx.x >> 2, tj = blockIdx.x & 3;
  const int il = threadIdx.x >> 4, jl = threadIdx.x & 15;
  float acc = 0.f;
  for (int s0 = 0; s0 < 2048; s0 += 64) {
    for (int t = threadIdx.x; t < 1024; t += 256) {
      const int ss = t >> 4, cc = t & 15;
      const float kv = Kf[(s0 + ss) * 1024 + h * 64 + ti * 16 + cc];
      skc[t] = kv > 0.f ? kv + 1.f : __expf(kv);
      vc[t]  = Vf[(s0 + ss) * 1024 + h * 64 + tj * 16 + cc];
    }
    __syncthreads();
#pragma unroll 16
    for (int ss = 0; ss < 64; ++ss) acc += skc[ss * 16 + il] * vc[ss * 16 + jl];
    __syncthreads();
  }
  const int i = ti * 16 + il, j = tj * 16 + jl;
  Mout[h * 4096 + i * 64 + j] = Min[h * 4096 + i * 64 + j] + acc;
}

__global__ void z_update(const float* __restrict__ Kf, const float* __restrict__ zin,
                         float* __restrict__ zout) {
  const int idx = blockIdx.x * blockDim.x + threadIdx.x;
  if (idx >= 1024) return;
  const int h = idx >> 6, i = idx & 63;
  float acc = 0.f;
  for (int s = 0; s < 2048; ++s) {
    const float kv = Kf[s * 1024 + h * 64 + i];
    acc += kv > 0.f ? kv + 1.f : __expf(kv);
  }
  zout[idx] = zin[idx] + acc;
}

extern "C" void kernel_launch(void* const* d_in, const int* in_sizes, int n_in,
                              void* d_out, int out_size, void* d_ws, size_t ws_size,
                              hipStream_t stream) {
  const float* x    = (const float*)d_in[0];
  const float* Min  = (const float*)d_in[1];
  const float* zin  = (const float*)d_in[2];
  const float* Wq   = (const float*)d_in[3];
  const float* Wk   = (const float*)d_in[4];
  const float* Wv   = (const float*)d_in[5];
  const float* Wo   = (const float*)d_in[6];
  const float* beta = (const float*)d_in[7];

  float* out  = (float*)d_out;        // [2048*1024]
  float* Mout = out + 2097152;        // [16*64*64]
  float* zout = Mout + 65536;         // [16*64]

  char* ws = (char*)d_ws;
  bf16_t* xp = (bf16_t*)(ws);                    // x packed A tiles      4 MiB
  bf16_t* wp = (bf16_t*)(ws + 4194304);          // Wq..Wo packed B tiles 8 MiB
  float*  Qf = (float*)(ws + 12582912);          // q f32                 8 MiB
  float*  Kf = (float*)(ws + 20971520);          // k f32                 8 MiB
  float*  Vf = (float*)(ws + 29360128);          // v f32                 8 MiB
  bf16_t* Qp = (bf16_t*)(ws + 37748736);         // q packed A (scaled)   4 MiB
  bf16_t* Kp = (bf16_t*)(ws + 41943040);         // k^T packed B          4 MiB
  bf16_t* Vp = (bf16_t*)(ws + 46137344);         // v packed B            4 MiB
  bf16_t* Op = (bf16_t*)(ws + 50331648);         // gated o packed A      4 MiB

  pack_x<<<256, 256, 0, stream>>>(x, xp);
  pack_w<<<256, 256, 0, stream>>>(Wq, wp);
  pack_w<<<256, 256, 0, stream>>>(Wk, wp + 1048576);
  pack_w<<<256, 256, 0, stream>>>(Wv, wp + 2097152);
  pack_w<<<256, 256, 0, stream>>>(Wo, wp + 3145728);

  dim3 gg(16, 16);
  wmma_gemm<<<gg, 256, 0, stream>>>(xp, wp,           Qf);
  wmma_gemm<<<gg, 256, 0, stream>>>(xp, wp + 1048576, Kf);
  wmma_gemm<<<gg, 256, 0, stream>>>(xp, wp + 2097152, Vf);

  pack_q<<<256, 256, 0, stream>>>(Qf, Qp);
  pack_kscore<<<256, 256, 0, stream>>>(Kf, Kp);
  pack_v<<<256, 256, 0, stream>>>(Vf, Vp);

  attn_kernel<<<dim3(16, 16), 256, 0, stream>>>(Qp, Kp, Vp, Qf, Min, zin, beta, Op);

  wmma_gemm<<<gg, 256, 0, stream>>>(Op, wp + 3145728, out);

  mem_update<<<dim3(16, 16), 256, 0, stream>>>(Kf, Vf, Min, Mout);
  z_update<<<4, 256, 0, stream>>>(Kf, zin, zout);
}
